// SSM_HIPPO_69922067578914
// MI455X (gfx1250) — matrix operand order repaired and checked by reading, not compiled, not run on hardware
//
#include <hip/hip_runtime.h>
#include <hip/hip_bf16.h>
#include <math.h>

// ---------- CDNA5 WMMA types ----------
typedef __attribute__((ext_vector_type(16))) __bf16 v16bf;
typedef __attribute__((ext_vector_type(8)))  __bf16 bf16x8;
typedef __attribute__((ext_vector_type(4)))  __bf16 bf16x4;
typedef __attribute__((ext_vector_type(8)))  float  v8f;
#define CAT16(lo,hi) __builtin_shufflevector(lo, hi, 0,1,2,3,4,5,6,7,8,9,10,11,12,13,14,15)

// ---------- model constants ----------
#define B_SZ      32
#define NUM_CH    24
#define SEQ_LEN   384
#define D_MODEL   512
#define NPATCH    47
#define PATCH_LEN 16
#define PSTRIDE   8
#define DT_RANK   32
#define FORECAST  96
#define NTOK      (B_SZ*NPATCH)   // 1504 tokens after patching

// ---------- GEMM tiling ----------
#define BM 64
#define BN 128
#define BK 32
#define LDP 40   // padded LDS row stride (bf16 elems): 80B, keeps 16B alignment

__device__ __forceinline__ float gelu_exact(float v) {
  return 0.5f * v * (1.0f + erff(v * 0.70710678118654752f));
}
__device__ __forceinline__ float silu(float v) {
  return v / (1.0f + expf(-v));
}

// C = A(MxK,f32,lda) * W(KxN,f32,ldw) [+bias] [+=C], bf16 WMMA, f32 accumulate.
// Requires: K % 32 == 0, N % 4 == 0, lda/ldw/ldc % 4 == 0, 16B-aligned pointers
// (true for every GEMM in this model). 8 waves = 2(M) x 4(N); each wave owns a
// 32x32 slab -> four v_wmma_f32_16x16x32_bf16 per K-step.
// A staged row-major [m][k]; W staged TRANSPOSED [n][k] via in-register 4x4
// transpose so every fragment is two contiguous ds_load_b128.
__global__ __launch_bounds__(256) void gemm_wmma_bf16(
    const float* __restrict__ A, int lda,
    const float* __restrict__ W, int ldw,
    const float* __restrict__ bias,
    float* __restrict__ C, int ldc,
    int M, int N, int K, int accum)
{
  __shared__ __bf16 As[BM][LDP];   // [m][k]  64x40  (5KB)
  __shared__ __bf16 Bs[BN][LDP];   // [n][k] 128x40 (10KB)
  const int tid    = threadIdx.x;
  const int wave   = tid >> 5;
  const int lane   = tid & 31;
  const int wm     = wave & 1;      // M-slab (0/32)
  const int wn     = wave >> 1;     // N-slab (0..3 -> *32)
  const int blockM = blockIdx.y * BM;
  const int blockN = blockIdx.x * BN;
  const int lrow   = lane & 15;
  const int khalf  = lane >> 4;

  // ---- hoisted A staging addresses: 2 x (row, 4 consecutive k) per thread ----
  const float* aP[2]; float aM[2]; __bf16* aS[2];
  #pragma unroll
  for (int u = 0; u < 2; ++u) {
    int idx = tid + u*256;          // 0..511 over 64x8 vec4 tiles
    int r = idx >> 3;
    int c = (idx & 7) << 2;
    int gr = blockM + r;
    int grc = (gr < M) ? gr : (M - 1);
    aM[u] = (gr < M) ? 1.f : 0.f;
    aP[u] = A + (size_t)grc*lda + c;
    aS[u] = &As[r][c];
  }
  // ---- hoisted W staging: one 4(k) x 4(n) block per thread ----
  const int wk = (tid >> 5) << 2;   // 0,4,..,28
  const int wc = (tid & 31) << 2;   // 0,4,..,124
  const int gc = blockN + wc;
  const int gcc = (gc < N) ? gc : 0;
  const float wMk = (gc < N) ? 1.f : 0.f;
  const float* wP = W + (size_t)wk*ldw + gcc;
  __bf16* bS = &Bs[wc][wk];

  // ---- hoisted fragment LDS pointers ----
  const __bf16* arow0 = &As[wm*32 +      lrow][0];
  const __bf16* arow1 = &As[wm*32 + 16 + lrow][0];
  const __bf16* brow0 = &Bs[wn*32 +      lrow][16*khalf];
  const __bf16* brow1 = &Bs[wn*32 + 16 + lrow][16*khalf];

  v8f c00 = {0.f,0.f,0.f,0.f,0.f,0.f,0.f,0.f};
  v8f c01 = c00, c10 = c00, c11 = c00;

  for (int k0 = 0; k0 < K; k0 += BK) {
    // unconditional vectorized global loads (mask applied at convert)
    float4 av0 = *(const float4*)(aP[0] + k0);
    float4 av1 = *(const float4*)(aP[1] + k0);
    float4 w0 = *(const float4*)(wP);
    float4 w1 = *(const float4*)(wP + ldw);
    float4 w2 = *(const float4*)(wP + 2*(size_t)ldw);
    float4 w3 = *(const float4*)(wP + 3*(size_t)ldw);
    wP += (size_t)BK * ldw;
    __syncthreads();                 // previous iteration's fragment reads done
    {
      bf16x4 p;
      p[0]=(__bf16)(av0.x*aM[0]); p[1]=(__bf16)(av0.y*aM[0]);
      p[2]=(__bf16)(av0.z*aM[0]); p[3]=(__bf16)(av0.w*aM[0]);
      *(bf16x4*)aS[0] = p;
      p[0]=(__bf16)(av1.x*aM[1]); p[1]=(__bf16)(av1.y*aM[1]);
      p[2]=(__bf16)(av1.z*aM[1]); p[3]=(__bf16)(av1.w*aM[1]);
      *(bf16x4*)aS[1] = p;
    }
    {
      bf16x4 t;   // 4x4 transpose in registers -> contiguous-k stores
      t[0]=(__bf16)(w0.x*wMk); t[1]=(__bf16)(w1.x*wMk); t[2]=(__bf16)(w2.x*wMk); t[3]=(__bf16)(w3.x*wMk);
      *(bf16x4*)(bS        ) = t;
      t[0]=(__bf16)(w0.y*wMk); t[1]=(__bf16)(w1.y*wMk); t[2]=(__bf16)(w2.y*wMk); t[3]=(__bf16)(w3.y*wMk);
      *(bf16x4*)(bS + LDP  ) = t;
      t[0]=(__bf16)(w0.z*wMk); t[1]=(__bf16)(w1.z*wMk); t[2]=(__bf16)(w2.z*wMk); t[3]=(__bf16)(w3.z*wMk);
      *(bf16x4*)(bS + 2*LDP) = t;
      t[0]=(__bf16)(w0.w*wMk); t[1]=(__bf16)(w1.w*wMk); t[2]=(__bf16)(w2.w*wMk); t[3]=(__bf16)(w3.w*wMk);
      *(bf16x4*)(bS + 3*LDP) = t;
    }
    __syncthreads();

    // fragments: A(16x32): lane -> row, k = j + 16*khalf (this code's contiguous order, B matches)
    v16bf a0 = CAT16(*(const bf16x8*)(arow0 + 16*khalf),
                     *(const bf16x8*)(arow0 + 16*khalf + 8));
    v16bf a1 = CAT16(*(const bf16x8*)(arow1 + 16*khalf),
                     *(const bf16x8*)(arow1 + 16*khalf + 8));
    // B(32x16): lane -> col, k = j + 16*khalf : contiguous in transposed tile
    v16bf b0 = CAT16(*(const bf16x8*)(brow0), *(const bf16x8*)(brow0 + 8));
    v16bf b1 = CAT16(*(const bf16x8*)(brow1), *(const bf16x8*)(brow1 + 8));

    c00 = __builtin_amdgcn_wmma_f32_16x16x32_bf16(false, a0, false, b0, (short)0, c00, false, false);
    c01 = __builtin_amdgcn_wmma_f32_16x16x32_bf16(false, a0, false, b1, (short)0, c01, false, false);
    c10 = __builtin_amdgcn_wmma_f32_16x16x32_bf16(false, a1, false, b0, (short)0, c10, false, false);
    c11 = __builtin_amdgcn_wmma_f32_16x16x32_bf16(false, a1, false, b1, (short)0, c11, false, false);
    __syncthreads();
  }

  // ---- epilogue; C/D layout: lane = N col, VGPR v -> M = v (+8 lanes 16..31)
  const int rb   = blockM + wm*32 + (khalf ? 8 : 0);
  const int col0 = blockN + wn*32 + lrow;
  const int col1 = col0 + 16;
  if (col0 < N) {
    const float bv = bias ? bias[col0] : 0.f;
    #pragma unroll
    for (int v = 0; v < 8; ++v) {
      int r0 = rb + v, r1 = rb + 16 + v;
      if (r0 < M) {
        float o = c00[v] + bv;
        if (accum) o += C[(size_t)r0*ldc + col0];
        C[(size_t)r0*ldc + col0] = o;
      }
      if (r1 < M) {
        float o = c10[v] + bv;
        if (accum) o += C[(size_t)r1*ldc + col0];
        C[(size_t)r1*ldc + col0] = o;
      }
    }
  }
  if (col1 < N) {
    const float bv = bias ? bias[col1] : 0.f;
    #pragma unroll
    for (int v = 0; v < 8; ++v) {
      int r0 = rb + v, r1 = rb + 16 + v;
      if (r0 < M) {
        float o = c01[v] + bv;
        if (accum) o += C[(size_t)r0*ldc + col1];
        C[(size_t)r0*ldc + col1] = o;
      }
      if (r1 < M) {
        float o = c11[v] + bv;
        if (accum) o += C[(size_t)r1*ldc + col1];
        C[(size_t)r1*ldc + col1] = o;
      }
    }
  }
}

// block-wide sum reduction (256 threads), leaves `red` reusable
__device__ __forceinline__ float block_sum(float v, float* red, int tid) {
  red[tid] = v; __syncthreads();
  for (int of = 128; of > 0; of >>= 1) {
    if (tid < of) red[tid] += red[tid + of];
    __syncthreads();
  }
  float r = red[0];
  __syncthreads();
  return r;
}

// Fused: h = gelu(LN(xp@W1 + b1)) + xp@Ws + bs  per sequence token (one block/token)
__global__ __launch_bounds__(256) void embed_kernel(
    const float* __restrict__ x,
    const float* __restrict__ W1, const float* __restrict__ b1,
    const float* __restrict__ g,  const float* __restrict__ be,
    const float* __restrict__ Ws, const float* __restrict__ bs,
    float* __restrict__ out)
{
  const int tok = blockIdx.x;                // b*SEQ + s
  const int b = tok / SEQ_LEN, s = tok % SEQ_LEN;
  __shared__ float xc[NUM_CH];
  __shared__ float red[256];
  const int tid = threadIdx.x;
  if (tid < NUM_CH) xc[tid] = x[(size_t)b*NUM_CH*SEQ_LEN + (size_t)tid*SEQ_LEN + s];
  __syncthreads();

  float t[2];
  #pragma unroll
  for (int ii = 0; ii < 2; ++ii) {
    int d = tid + ii*256;
    float acc = b1[d];
    for (int c = 0; c < NUM_CH; ++c) acc += xc[c] * W1[c*D_MODEL + d];
    t[ii] = acc;
  }
  float m = block_sum(t[0] + t[1], red, tid) * (1.0f/(float)D_MODEL);
  float d0 = t[0]-m, d1 = t[1]-m;
  float var = block_sum(d0*d0 + d1*d1, red, tid) * (1.0f/(float)D_MODEL);
  float rs = rsqrtf(var + 1e-5f);
  #pragma unroll
  for (int ii = 0; ii < 2; ++ii) {
    int d = tid + ii*256;
    float v  = (t[ii]-m)*rs*g[d] + be[d];
    float ge = gelu_exact(v);
    float sk = bs[d];
    for (int c = 0; c < NUM_CH; ++c) sk += xc[c] * Ws[c*D_MODEL + d];
    out[(size_t)tok*D_MODEL + d] = ge + sk;
  }
}

// hp[b,p,d] = mean_j hseq[b, p*8+j, d]
__global__ void patch_kernel(const float* __restrict__ hseq, float* __restrict__ out, int total) {
  int i = blockIdx.x*blockDim.x + threadIdx.x;
  if (i >= total) return;
  int d = i % D_MODEL; int rest = i / D_MODEL;
  int p = rest % NPATCH; int b = rest / NPATCH;
  float acc = 0.f;
  #pragma unroll
  for (int j = 0; j < PATCH_LEN; ++j)
    acc += hseq[((size_t)b*SEQ_LEN + p*PSTRIDE + j)*D_MODEL + d];
  out[i] = acc * (1.0f/(float)PATCH_LEN);
}

// out = gelu(LN(g1; gain,beta)) [+ skipbuf]   one block per token, dim in {512,1024}
__global__ __launch_bounds__(256) void ln_gelu_add_kernel(
    const float* __restrict__ g1, const float* __restrict__ gain,
    const float* __restrict__ beta, const float* __restrict__ skipb,
    float* __restrict__ out, int dim)
{
  const int tok = blockIdx.x, tid = threadIdx.x;
  __shared__ float red[256];
  const float* row = g1 + (size_t)tok*dim;
  const int cnt = dim >> 8;                 // 2 or 4
  float vals[4];
  float sum = 0.f;
  for (int ii = 0; ii < cnt; ++ii) { vals[ii] = row[tid + ii*256]; sum += vals[ii]; }
  float m = block_sum(sum, red, tid) / (float)dim;
  float s2 = 0.f;
  for (int ii = 0; ii < cnt; ++ii) { float dd = vals[ii]-m; s2 += dd*dd; }
  float var = block_sum(s2, red, tid) / (float)dim;
  float rs = rsqrtf(var + 1e-5f);
  for (int ii = 0; ii < cnt; ++ii) {
    int d = tid + ii*256;
    float v  = (vals[ii]-m)*rs*gain[d] + beta[d];
    float ge = gelu_exact(v);
    if (skipb) ge += skipb[(size_t)tok*dim + d];
    out[(size_t)tok*dim + d] = ge;
  }
}

// causal depthwise conv (D_CONV=4, left pad 3) + bias + SiLU over patch axis
__global__ void conv_silu_kernel(const float* __restrict__ xmres,
                                 const float* __restrict__ convW,
                                 const float* __restrict__ convB,
                                 float* __restrict__ xmact, int di)
{
  int i = blockIdx.x*blockDim.x + threadIdx.x;
  if (i >= B_SZ*di) return;
  int b = i / di, d = i % di;
  const size_t base = (size_t)b*NPATCH;
  const size_t str  = (size_t)2*di;
  float w0 = convW[d*4+0], w1 = convW[d*4+1], w2 = convW[d*4+2], w3 = convW[d*4+3];
  float cb = convB[d];
  for (int p = 0; p < NPATCH; ++p) {
    float acc = cb;
    int t;
    t = p-3; if (t >= 0) acc += w0 * xmres[(base+t)*str + d];
    t = p-2; if (t >= 0) acc += w1 * xmres[(base+t)*str + d];
    t = p-1; if (t >= 0) acc += w2 * xmres[(base+t)*str + d];
    acc += w3 * xmres[(base+p)*str + d];
    xmact[((size_t)b*NPATCH + p)*di + d] = silu(acc);
  }
}

__global__ void softplus_kernel(float* __restrict__ x, long total) {
  long i = (long)blockIdx.x*blockDim.x + threadIdx.x;
  if (i >= total) return;
  float v = x[i];
  x[i] = (v > 20.f) ? v : log1pf(expf(v));
}

// selective scan; A_h (HIPPO-LegS) built in LDS, B_h per-thread. one thread = (b,d)
__global__ __launch_bounds__(256) void scan_kernel(
    const float* __restrict__ xmact, const float* __restrict__ dlt,
    const float* __restrict__ Cmat, int ldc,
    float* __restrict__ y, int di, int n)
{
  __shared__ float A[32*32];
  const int tid = threadIdx.x;
  for (int i = tid; i < n*n; i += 256) {
    int m = i / n, j = i % n;
    float v;
    if (m > j)       v = -sqrtf((2.f*m+1.f)*(2.f*j+1.f));
    else if (m == j) v = -((float)m + 1.f);
    else             v = 0.f;
    A[i] = v;
  }
  __syncthreads();
  int gid = blockIdx.x*256 + tid;
  if (gid >= B_SZ*di) return;
  int b = gid / di, d = gid % di;
  float s[32], bh[32];
  for (int j = 0; j < n; ++j) { s[j] = 0.f; bh[j] = sqrtf(2.f*j + 1.f); }
  for (int p = 0; p < NPATCH; ++p) {
    size_t row = (size_t)b*NPATCH + p;
    float dt = dlt[row*di + d];
    float u  = xmact[row*di + d];
    float t[32];
    for (int m = 0; m < n; ++m) {
      float acc = 0.f;
      for (int j = 0; j < n; ++j) acc += A[m*n + j] * s[j];
      t[m] = s[m] + dt * (acc + u * bh[m]);
    }
    const float* Crow = Cmat + row*(size_t)ldc;
    float yv = 0.f;
    for (int j = 0; j < n; ++j) { s[j] = t[j]; yv += t[j] * Crow[j]; }
    y[row*di + d] = yv;
  }
}

// y = (y + xm*D) * silu(res)
__global__ void ypost_kernel(float* __restrict__ y, const float* __restrict__ xmact,
                             const float* __restrict__ xmres, const float* __restrict__ Dp,
                             int di, long total)
{
  long i = (long)blockIdx.x*blockDim.x + threadIdx.x;
  if (i >= total) return;
  int d = (int)(i % di);
  long row = i / di;
  float res = xmres[(size_t)row*(size_t)(2*di) + di + d];
  y[i] = (y[i] + xmact[i]*Dp[d]) * silu(res);
}

// out = h * rsqrt(mean(h^2)+eps) * w ; one block per token
__global__ __launch_bounds__(256) void rmsnorm_kernel(
    const float* __restrict__ h, const float* __restrict__ w,
    float* __restrict__ out, int dim)
{
  const int tok = blockIdx.x, tid = threadIdx.x;
  __shared__ float red[256];
  const float* row = h + (size_t)tok*dim;
  float acc = 0.f;
  for (int d = tid; d < dim; d += 256) { float v = row[d]; acc += v*v; }
  float ms = block_sum(acc, red, tid) / (float)dim;
  float rs = rsqrtf(ms + 1e-5f);
  for (int d = tid; d < dim; d += 256)
    out[(size_t)tok*dim + d] = row[d] * rs * w[d];
}

extern "C" void kernel_launch(void* const* d_in, const int* in_sizes, int n_in,
                              void* d_out, int out_size, void* d_ws, size_t ws_size,
                              hipStream_t stream)
{
  (void)in_sizes; (void)out_size; (void)ws_size;
  if (n_in < 64) return;
  int ix = 0;
  auto nxt = [&]() { return (const float*)d_in[ix++]; };
  const float* x         = nxt();
  const float* in_proj_W = nxt(); const float* in_proj_b = nxt();
  const float* in_ln_g   = nxt(); const float* in_ln_b   = nxt();
  const float* g_skip_W  = nxt(); const float* g_skip_b  = nxt();
  const float *mainW[4], *mainB[4], *mainG[4], *mainBe[4];
  const float *mbW[4], *convW[4], *convB[4], *xW[4], *dtW[4], *dtB[4], *Dp[4], *outW[4];
  const float *skW[4] = {nullptr,nullptr,nullptr,nullptr};
  const float *skB[4] = {nullptr,nullptr,nullptr,nullptr};
  for (int l = 0; l < 4; ++l) {
    mainW[l]=nxt(); mainB[l]=nxt(); mainG[l]=nxt(); mainBe[l]=nxt();
    mbW[l]=nxt();   convW[l]=nxt(); convB[l]=nxt(); xW[l]=nxt();
    dtW[l]=nxt();   dtB[l]=nxt();   Dp[l]=nxt();    outW[l]=nxt();
    if (l > 0) { skW[l]=nxt(); skB[l]=nxt(); }
  }
  const float* norm_w = nxt();
  const float* head_W = nxt();
  const float* head_b = nxt();

  // workspace layout (f32), ~118 MB total
  float* ws = (float*)d_ws;
  size_t o = 0;
  float* hseq  = ws + o; o += (size_t)B_SZ*SEQ_LEN*D_MODEL;
  float* hA    = ws + o; o += (size_t)NTOK*1024;
  float* hB    = ws + o; o += (size_t)NTOK*1024;
  float* g1    = ws + o; o += (size_t)NTOK*1024;
  float* skb   = ws + o; o += (size_t)NTOK*1024;
  float* xmres = ws + o; o += (size_t)NTOK*4096;
  float* xmact = ws + o; o += (size_t)NTOK*2048;
  float* dblb  = ws + o; o += (size_t)NTOK*96;
  float* dlt   = ws + o; o += (size_t)NTOK*2048;
  float* ybuf  = ws + o; o += (size_t)NTOK*2048;
  float* hn    = ws + o; o += (size_t)NTOK*1024;

  embed_kernel<<<B_SZ*SEQ_LEN, 256, 0, stream>>>(
      x, in_proj_W, in_proj_b, in_ln_g, in_ln_b, g_skip_W, g_skip_b, hseq);
  {
    int total = B_SZ*NPATCH*D_MODEL;
    patch_kernel<<<(total+255)/256, 256, 0, stream>>>(hseq, hA, total);
  }

  const int DIN[4]  = {512, 512, 512, 1024};
  const int DOUT[4] = {512, 512, 1024, 1024};
  const int NSx[4]  = {16, 16, 32, 32};

  float* Hin = hA; float* Hout = hB;
  for (int l = 0; l < 4; ++l) {
    const int din = DIN[l], dout = DOUT[l], n = NSx[l], di = 2*dout;
    dim3 gM((dout+BN-1)/BN, (NTOK+BM-1)/BM);
    // main proj (+bias)
    gemm_wmma_bf16<<<gM, 256, 0, stream>>>(Hin, din, mainW[l], dout, mainB[l],
                                           g1, dout, NTOK, dout, din, 0);
    // layer skip proj (+bias)
    if (l > 0)
      gemm_wmma_bf16<<<gM, 256, 0, stream>>>(Hin, din, skW[l], dout, skB[l],
                                             skb, dout, NTOK, dout, din, 0);
    // LN + GELU (+skip) -> Hout
    ln_gelu_add_kernel<<<NTOK, 256, 0, stream>>>(g1, mainG[l], mainBe[l],
                                                 (l>0)?skb:nullptr, Hout, dout);
    // mamba in-proj -> [xm | res]
    dim3 gmb((2*di+BN-1)/BN, (NTOK+BM-1)/BM);
    gemm_wmma_bf16<<<gmb, 256, 0, stream>>>(Hout, dout, mbW[l], 2*di, nullptr,
                                            xmres, 2*di, NTOK, 2*di, dout, 0);
    // depthwise causal conv + SiLU
    conv_silu_kernel<<<(B_SZ*di+255)/256, 256, 0, stream>>>(xmres, convW[l], convB[l], xmact, di);
    // x-proj -> [dt_rank | B | C]
    const int nd = DT_RANK + 2*n;
    dim3 gdb((nd+BN-1)/BN, (NTOK+BM-1)/BM);
    gemm_wmma_bf16<<<gdb, 256, 0, stream>>>(xmact, di, xW[l], nd, nullptr,
                                            dblb, nd, NTOK, nd, di, 0);
    // delta = softplus(dbl[:, :32] @ dt_W + dt_b)
    dim3 gdt((di+BN-1)/BN, (NTOK+BM-1)/BM);
    gemm_wmma_bf16<<<gdt, 256, 0, stream>>>(dblb, nd, dtW[l], di, dtB[l],
                                            dlt, di, NTOK, di, DT_RANK, 0);
    long tot = (long)NTOK*di;
    softplus_kernel<<<(int)((tot+255)/256), 256, 0, stream>>>(dlt, tot);
    // selective scan (C slice starts at col 32+n, stride nd)
    scan_kernel<<<(B_SZ*di+255)/256, 256, 0, stream>>>(xmact, dlt,
                                                       dblb + DT_RANK + n, nd, ybuf, di, n);
    // gate: y = (y + xm*D) * silu(res)
    ypost_kernel<<<(int)((tot+255)/256), 256, 0, stream>>>(ybuf, xmact, xmres, Dp[l], di, tot);
    // out proj, accumulated into residual stream: Hout += y @ out_W
    gemm_wmma_bf16<<<gM, 256, 0, stream>>>(ybuf, di, outW[l], dout, nullptr,
                                           Hout, dout, NTOK, dout, di, 1);
    float* t = Hin; Hin = Hout; Hout = t;
  }

  // RMSNorm + head
  rmsnorm_kernel<<<NTOK, 256, 0, stream>>>(Hin, norm_w, hn, 1024);
  dim3 gh((NUM_CH*FORECAST+BN-1)/BN, (B_SZ+BM-1)/BM);
  gemm_wmma_bf16<<<gh, 256, 0, stream>>>(hn, NPATCH*1024, head_W, NUM_CH*FORECAST, head_b,
                                         (float*)d_out, NUM_CH*FORECAST,
                                         B_SZ, NUM_CH*FORECAST, NPATCH*1024, 0);
}
